// SelectiveSSM_76467597738247
// MI455X (gfx1250) — matrix operand order repaired and checked by reading, not compiled, not run on hardware
//
#include <hip/hip_runtime.h>
#include <hip/hip_bf16.h>

// ---------------------------------------------------------------------------
// Selective SSM forward for gfx1250 (MI455X):
//   - bf16 WMMA GEMMs (v_wmma_f32_16x16x32_bf16)
//   - chunk-parallel linear-recurrence scan (64 chunks x 64 steps)
//   - TDM (tensor_load_to_lds) double-buffered LDS staging for the big GEMM
// ---------------------------------------------------------------------------

typedef __bf16 bf16;
typedef __attribute__((ext_vector_type(16))) __bf16 v16bf;
typedef __attribute__((ext_vector_type(8)))  __bf16 v8bf;
typedef __attribute__((ext_vector_type(8)))  float  v8f;
typedef __attribute__((ext_vector_type(4)))  unsigned v4u;
typedef __attribute__((ext_vector_type(8)))  unsigned v8u;

#define DIM    1024
#define NST    64
#define SEQ    4096
#define BATCH  2
#define MROWS  (BATCH * SEQ)   // 8192 token rows
#define PCOLS  192             // dt_rank + 2*n_state
#define CHUNKS 64              // scan chunks
#define CT     (SEQ / CHUNKS)  // 64 steps per chunk
#define KC     64              // GEMM3 k-chunk staged via TDM
#define LDSROW 72              // bf16 elems per LDS tile row (64 + 8 pad = 144B)

__device__ __forceinline__ bf16 f2bf(float f) {
  unsigned u = __builtin_bit_cast(unsigned, f);
  u += 0x7FFFu + ((u >> 16) & 1u);              // round-to-nearest-even
  unsigned short h = (unsigned short)(u >> 16);
  return __builtin_bit_cast(bf16, h);
}

// ---------------------------------------------------------------------------
// Weight convert + transpose: in f32 (K x N row-major) -> out bf16 (N x K).
// ---------------------------------------------------------------------------
__global__ void cvtT_kernel(const float* __restrict__ in, bf16* __restrict__ out,
                            int K, int N) {
  int idx = blockIdx.x * blockDim.x + threadIdx.x;
  if (idx >= K * N) return;
  int k = idx / N, n = idx % N;
  out[(size_t)n * K + k] = f2bf(in[idx]);
}

// ---------------------------------------------------------------------------
// LayerNorm over D=1024, one block per (b,l) row.
// ---------------------------------------------------------------------------
__global__ __launch_bounds__(256) void layernorm_kernel(
    const float* __restrict__ x, const float* __restrict__ w,
    const float* __restrict__ b, float* __restrict__ xn, bf16* __restrict__ xnb) {
  __shared__ float s1[256];
  __shared__ float s2[256];
  const size_t base = (size_t)blockIdx.x * DIM;
  float v[4];
  float s = 0.f, q = 0.f;
#pragma unroll
  for (int j = 0; j < 4; ++j) {
    v[j] = x[base + threadIdx.x + j * 256];
    s += v[j];
    q += v[j] * v[j];
  }
  s1[threadIdx.x] = s;
  s2[threadIdx.x] = q;
  __syncthreads();
  for (int off = 128; off > 0; off >>= 1) {
    if (threadIdx.x < (unsigned)off) {
      s1[threadIdx.x] += s1[threadIdx.x + off];
      s2[threadIdx.x] += s2[threadIdx.x + off];
    }
    __syncthreads();
  }
  const float mu   = s1[0] * (1.f / DIM);
  const float var  = s2[0] * (1.f / DIM) - mu * mu;
  const float rstd = rsqrtf(var + 1e-5f);
#pragma unroll
  for (int j = 0; j < 4; ++j) {
    const int idx = threadIdx.x + j * 256;
    const float o = (v[j] - mu) * rstd * w[idx] + b[idx];
    xn[base + idx]  = o;
    xnb[base + idx] = f2bf(o);
  }
}

// ---------------------------------------------------------------------------
// WMMA strip core (A and B straight from global): one wave does 16(M)x64(N).
// Fragment k order: contiguous, k = 16*half + s, same for A and B (wave32).
// ---------------------------------------------------------------------------
__device__ __forceinline__ void wmma_strip(const bf16* __restrict__ A,
                                           const bf16* __restrict__ Bt,
                                           int K, int m0, int n0, v8f c[4]) {
  const int lane = threadIdx.x & 31;
  const int rc   = lane & 15;
  const int half = lane >> 4;
  const bf16* arow = A + (size_t)(m0 + rc) * K;
  for (int kb = 0; kb < K; kb += 32) {
    v8bf alo = *(const v8bf*)(arow + kb + 16 * half);
    v8bf ahi = *(const v8bf*)(arow + kb + 16 * half + 8);
    v16bf a = __builtin_shufflevector(alo, ahi,
        0, 1, 2, 3, 4, 5, 6, 7, 8, 9, 10, 11, 12, 13, 14, 15);
#pragma unroll
    for (int t = 0; t < 4; ++t) {
      const bf16* brow = Bt + (size_t)(n0 + t * 16 + rc) * K + kb + 16 * half;
      v8bf b0 = *(const v8bf*)(brow);
      v8bf b1 = *(const v8bf*)(brow + 8);
      v16bf bm = __builtin_shufflevector(b0, b1,
          0, 1, 2, 3, 4, 5, 6, 7, 8, 9, 10, 11, 12, 13, 14, 15);
      c[t] = __builtin_amdgcn_wmma_f32_16x16x32_bf16(
          false, a, false, bm, (short)0, c[t], false, false);
    }
  }
}

// GEMM1: proj = xn @ w_xproj  (8192 x 1024 x 192); g==0 also emits dtr bf16.
__global__ __launch_bounds__(32) void gemm_proj_kernel(
    const bf16* __restrict__ xnb, const bf16* __restrict__ wT,
    float* __restrict__ proj, bf16* __restrict__ dtr) {
  const int m0 = blockIdx.x * 16;
  const int g  = blockIdx.y;
  const int n0 = g * 64;
  v8f c[4] = {};
  wmma_strip(xnb, wT, DIM, m0, n0, c);
  const int lane = threadIdx.x & 31;
  const int col  = lane & 15;
  const int half = lane >> 4;
#pragma unroll
  for (int t = 0; t < 4; ++t) {
    const int n = n0 + t * 16 + col;
#pragma unroll
    for (int j = 0; j < 8; ++j) {
      const int m = m0 + j + 8 * half;
      const float v = c[t][j];
      proj[(size_t)m * PCOLS + n] = v;
      if (g == 0) dtr[(size_t)m * 64 + n] = f2bf(v);
    }
  }
}

// GEMM2: delta = softplus(dtr @ w_dt + b_dt)  (8192 x 64 x 1024).
__global__ __launch_bounds__(32) void gemm_delta_kernel(
    const bf16* __restrict__ dtr, const bf16* __restrict__ wT,
    const float* __restrict__ bias, float* __restrict__ delta) {
  const int m0 = blockIdx.x * 16;
  const int n0 = blockIdx.y * 64;
  v8f c[4] = {};
  wmma_strip(dtr, wT, 64, m0, n0, c);
  const int lane = threadIdx.x & 31;
  const int col  = lane & 15;
  const int half = lane >> 4;
#pragma unroll
  for (int t = 0; t < 4; ++t) {
    const int n = n0 + t * 16 + col;
    const float bn = bias[n];
#pragma unroll
    for (int j = 0; j < 8; ++j) {
      const int m = m0 + j + 8 * half;
      const float v = c[t][j] + bn;
      const float sp = fmaxf(v, 0.f) + log1pf(__expf(-fabsf(v)));
      delta[(size_t)m * DIM + n] = sp;
    }
  }
}

// ---------------------------------------------------------------------------
// TDM issue: DMA a 64-row x 64-col bf16 tile (row stride = DIM elems) from
// global into LDS, padding each 128B row with 16B (pad_interval=32 dwords,
// pad_amount=4 dwords) -> 144B LDS row stride, bank-conflict-free fragments.
// D# layout per CDNA5 ISA.
// ---------------------------------------------------------------------------
__device__ __forceinline__ void tdm_load_tile(unsigned long long gaddr,
                                              unsigned lds_addr) {
  const v4u g0 = {
      1u,                                                      // count=1 (user)
      lds_addr,                                                // lds_addr bytes
      (unsigned)gaddr,                                         // global_addr lo
      ((unsigned)(gaddr >> 32) & 0x01FFFFFFu) | (2u << 30)     // addr hi | type=2
  };
  const v8u g1 = {
      (1u << 16) | (1u << 20) | (4u << 22) | (3u << 25),  // data_size=2B, pad: 32dw+4dw
      ((unsigned)DIM & 0xFFFFu) << 16,                    // tensor_dim0 = K
      (64u << 16),                                        // tensor_dim1 = 64 rows
      (64u << 16),                                        // tile_dim0 = 64
      64u,                                                // tile_dim1 = 64
      (unsigned)DIM,                                      // tensor_dim0_stride = K
      0u, 0u
  };
  asm volatile("tensor_load_to_lds %0, %1" :: "s"(g0), "s"(g1) : "memory");
}

// GEMM3: out = z @ w_out  (8192 x 1024 x 1024).
// 128-thread block = 4 waves sharing a TDM-staged, double-buffered B tile.
__global__ __launch_bounds__(128) void gemm_out_kernel(
    const bf16* __restrict__ zb, const bf16* __restrict__ wT,
    float* __restrict__ out) {
  __shared__ bf16 tile[2][64 * LDSROW];     // 2 x 9216 B
  const int wave = threadIdx.x >> 5;
  const int lane = threadIdx.x & 31;
  const int rc   = lane & 15;
  const int half = lane >> 4;
  const int m0   = blockIdx.x * 64 + wave * 16;
  const int n0   = blockIdx.y * 64;

  const unsigned long long gbase =
      (unsigned long long)(const void*)(wT + (size_t)n0 * DIM);
  const unsigned lds0 = (unsigned)(size_t)(void*)&tile[0][0];
  const unsigned lds1 = (unsigned)(size_t)(void*)&tile[1][0];

  v8f c[4] = {};
  const bf16* arow = zb + (size_t)(m0 + rc) * DIM;

  if (wave == 0) tdm_load_tile(gbase, lds0);

  for (int ch = 0; ch < DIM / KC; ++ch) {           // 16 k-chunks
    if (wave == 0) {
      if (ch + 1 < DIM / KC) {
        tdm_load_tile(gbase + (unsigned long long)(ch + 1) * KC * sizeof(bf16),
                      ((ch + 1) & 1) ? lds1 : lds0);
        __builtin_amdgcn_s_wait_tensorcnt(1);       // current chunk landed
      } else {
        __builtin_amdgcn_s_wait_tensorcnt(0);       // last chunk landed
      }
    }
    __syncthreads();                                // tile[ch&1] ready for all waves

    const bf16* tb = &tile[ch & 1][0];
    const int kb0 = ch * KC;
#pragma unroll
    for (int ks = 0; ks < KC; ks += 32) {
      const int kb = kb0 + ks;
      v8bf alo = *(const v8bf*)(arow + kb + 16 * half);
      v8bf ahi = *(const v8bf*)(arow + kb + 16 * half + 8);
      v16bf a = __builtin_shufflevector(alo, ahi,
          0, 1, 2, 3, 4, 5, 6, 7, 8, 9, 10, 11, 12, 13, 14, 15);
#pragma unroll
      for (int t = 0; t < 4; ++t) {
        const bf16* brow = tb + (size_t)(t * 16 + rc) * LDSROW + ks + 16 * half;
        v8bf b0 = *(const v8bf*)(brow);             // ds_load_b128
        v8bf b1 = *(const v8bf*)(brow + 8);
        v16bf bm = __builtin_shufflevector(b0, b1,
            0, 1, 2, 3, 4, 5, 6, 7, 8, 9, 10, 11, 12, 13, 14, 15);
        c[t] = __builtin_amdgcn_wmma_f32_16x16x32_bf16(
            false, a, false, bm, (short)0, c[t], false, false);
      }
    }
    __syncthreads();                                // all done reading tile[ch&1]
  }

#pragma unroll
  for (int t = 0; t < 4; ++t) {
    const int n = n0 + t * 16 + rc;
#pragma unroll
    for (int j = 0; j < 8; ++j) {
      const int m = m0 + j + 8 * half;
      out[(size_t)m * DIM + n] = c[t][j];
    }
  }
}

// ---------------------------------------------------------------------------
// Chunk-parallel scan. Recurrence h <- exp(dt*A)*h + dt*x*B is linear in h:
//   chunk decay = Prod exp(dt_l*A[n]) = exp(A[n] * Sum dt_l)  (exact)
// Phase A: per chunk, local terminal state hT (h0=0) and S = Sum dt.
// Phase B: compose 64 chunk boundaries serially -> per-chunk entry state h0.
// Phase C: re-run each chunk from its entry state, emit z = y + x*Dp (bf16).
// ---------------------------------------------------------------------------
__global__ __launch_bounds__(64) void scan_partial_kernel(
    const float* __restrict__ delta, const float* __restrict__ xn,
    const float* __restrict__ proj,  const float* __restrict__ A_log,
    float* __restrict__ hT, float* __restrict__ S) {
  int id = blockIdx.x;
  const int dblk = id & 15; id >>= 4;
  const int c    = id & (CHUNKS - 1); id >>= 6;
  const int b    = id;
  const int d    = dblk * 64 + threadIdx.x;

  float A[NST];
#pragma unroll
  for (int n = 0; n < NST; ++n) A[n] = -__expf(A_log[(size_t)d * NST + n]);

  float h[NST];
#pragma unroll
  for (int n = 0; n < NST; ++n) h[n] = 0.f;
  float s = 0.f;

  const int l0 = c * CT;
  for (int l = l0; l < l0 + CT; ++l) {
    const size_t r = (size_t)b * SEQ + l;
    const float dt = delta[r * DIM + d];
    const float xv = xn[r * DIM + d];
    const float* Bl = proj + r * PCOLS + 64;      // wave-uniform -> s_load
    if (l + 1 < l0 + CT) {
      __builtin_prefetch(delta + (r + 1) * DIM + d, 0, 1);
      __builtin_prefetch(proj + (r + 1) * PCOLS + 64, 0, 1);
    }
    s += dt;
    const float dtx = dt * xv;
#pragma unroll
    for (int n = 0; n < NST; ++n)
      h[n] = fmaf(__expf(dt * A[n]), h[n], dtx * Bl[n]);
  }

  const size_t bc = (size_t)b * CHUNKS + c;
  S[bc * DIM + d] = s;
#pragma unroll
  for (int n = 0; n < NST; ++n)
    hT[(bc * NST + n) * DIM + d] = h[n];          // [b][c][n][d], d coalesced
}

__global__ __launch_bounds__(256) void scan_stitch_kernel(
    const float* __restrict__ hT, const float* __restrict__ S,
    const float* __restrict__ A_log, float* __restrict__ h0) {
  const int g = blockIdx.x * 256 + threadIdx.x;   // over B*NST*DIM
  const int d = g % DIM;
  const int n = (g / DIM) % NST;
  const int b = g / (NST * DIM);
  const float An = -__expf(A_log[(size_t)d * NST + n]);
  float h = 0.f;
  for (int c = 0; c < CHUNKS; ++c) {
    const size_t bc = (size_t)b * CHUNKS + c;
    h0[(bc * NST + n) * DIM + d] = h;             // entry state of chunk c
    const float P = __expf(An * S[bc * DIM + d]); // chunk decay product
    h = fmaf(P, h, hT[(bc * NST + n) * DIM + d]);
  }
}

__global__ __launch_bounds__(64) void scan_apply_kernel(
    const float* __restrict__ delta, const float* __restrict__ xn,
    const float* __restrict__ x,     const float* __restrict__ proj,
    const float* __restrict__ A_log, const float* __restrict__ Dp,
    const float* __restrict__ h0,    bf16* __restrict__ zb) {
  int id = blockIdx.x;
  const int dblk = id & 15; id >>= 4;
  const int c    = id & (CHUNKS - 1); id >>= 6;
  const int b    = id;
  const int d    = dblk * 64 + threadIdx.x;

  float A[NST];
#pragma unroll
  for (int n = 0; n < NST; ++n) A[n] = -__expf(A_log[(size_t)d * NST + n]);
  const float dpd = Dp[d];

  const size_t bc = (size_t)b * CHUNKS + c;
  float h[NST];
#pragma unroll
  for (int n = 0; n < NST; ++n) h[n] = h0[(bc * NST + n) * DIM + d];

  const int l0 = c * CT;
  for (int l = l0; l < l0 + CT; ++l) {
    const size_t r = (size_t)b * SEQ + l;
    const float dt = delta[r * DIM + d];
    const float xv = xn[r * DIM + d];
    const float xo = x[r * DIM + d];
    const float* Bl = proj + r * PCOLS + 64;      // wave-uniform
    const float* Cl = Bl + 64;
    if (l + 1 < l0 + CT) {
      __builtin_prefetch(delta + (r + 1) * DIM + d, 0, 1);
      __builtin_prefetch(proj + (r + 1) * PCOLS + 64, 0, 1);
    }
    const float dtx = dt * xv;
    float y = 0.f;
#pragma unroll
    for (int n = 0; n < NST; ++n) {
      h[n] = fmaf(__expf(dt * A[n]), h[n], dtx * Bl[n]);
      y = fmaf(h[n], Cl[n], y);
    }
    zb[r * DIM + d] = f2bf(y + xo * dpd);
  }
}

// ---------------------------------------------------------------------------
extern "C" void kernel_launch(void* const* d_in, const int* in_sizes, int n_in,
                              void* d_out, int out_size, void* d_ws, size_t ws_size,
                              hipStream_t stream) {
  const float* x       = (const float*)d_in[0];
  const float* ln_w    = (const float*)d_in[1];
  const float* ln_b    = (const float*)d_in[2];
  const float* w_xproj = (const float*)d_in[3];
  const float* w_dt    = (const float*)d_in[4];
  const float* b_dt    = (const float*)d_in[5];
  const float* A_log   = (const float*)d_in[6];
  const float* Dp      = (const float*)d_in[7];
  const float* w_out   = (const float*)d_in[8];

  char* ws = (char*)d_ws;
  size_t off = 0;
  auto carve = [&](size_t bytes) -> void* {
    void* p = ws + off;
    off = (off + bytes + 255) & ~(size_t)255;
    return p;
  };
  bf16*  wxpT  = (bf16*) carve((size_t)PCOLS * DIM * 2);
  bf16*  wdtT  = (bf16*) carve((size_t)DIM * 64 * 2);
  bf16*  woutT = (bf16*) carve((size_t)DIM * DIM * 2);
  float* xn    = (float*)carve((size_t)MROWS * DIM * 4);
  bf16*  xnb   = (bf16*) carve((size_t)MROWS * DIM * 2);
  float* proj  = (float*)carve((size_t)MROWS * PCOLS * 4);
  bf16*  dtr   = (bf16*) carve((size_t)MROWS * 64 * 2);
  float* delta = (float*)carve((size_t)MROWS * DIM * 4);
  bf16*  zb    = (bf16*) carve((size_t)MROWS * DIM * 2);
  float* hT    = (float*)carve((size_t)BATCH * CHUNKS * NST * DIM * 4); // 32 MB
  float* h0    = (float*)carve((size_t)BATCH * CHUNKS * NST * DIM * 4); // 32 MB
  float* Ssum  = (float*)carve((size_t)BATCH * CHUNKS * DIM * 4);       // 512 KB
  (void)in_sizes; (void)n_in; (void)out_size; (void)ws_size;

  cvtT_kernel<<<(DIM * PCOLS + 255) / 256, 256, 0, stream>>>(w_xproj, wxpT, DIM, PCOLS);
  cvtT_kernel<<<(64 * DIM + 255) / 256, 256, 0, stream>>>(w_dt, wdtT, 64, DIM);
  cvtT_kernel<<<(DIM * DIM + 255) / 256, 256, 0, stream>>>(w_out, woutT, DIM, DIM);

  layernorm_kernel<<<MROWS, 256, 0, stream>>>(x, ln_w, ln_b, xn, xnb);
  gemm_proj_kernel<<<dim3(MROWS / 16, 3), 32, 0, stream>>>(xnb, wxpT, proj, dtr);
  gemm_delta_kernel<<<dim3(MROWS / 16, DIM / 64), 32, 0, stream>>>(dtr, wdtT, b_dt, delta);

  const int scan_blocks = BATCH * CHUNKS * (DIM / 64);   // 2048
  scan_partial_kernel<<<scan_blocks, 64, 0, stream>>>(delta, xn, proj, A_log, hT, Ssum);
  scan_stitch_kernel<<<(BATCH * NST * DIM) / 256, 256, 0, stream>>>(hT, Ssum, A_log, h0);
  scan_apply_kernel<<<scan_blocks, 64, 0, stream>>>(delta, xn, x, proj, A_log, Dp, h0, zb);

  gemm_out_kernel<<<dim3(MROWS / 64, DIM / 64), 128, 0, stream>>>(zb, woutT, (float*)d_out);
}
